// RNN_Net_8435315769736
// MI455X (gfx1250) — matrix operand order repaired and checked by reading, not compiled, not run on hardware
//
#include <hip/hip_runtime.h>
#include <hip/hip_bf16.h>

typedef __attribute__((ext_vector_type(2)))  float  f32x2;
typedef __attribute__((ext_vector_type(4)))  float  f32x4;
typedef __attribute__((ext_vector_type(8)))  float  f32x8;
typedef __attribute__((ext_vector_type(4)))  __bf16 bf16x4;
typedef __attribute__((ext_vector_type(8)))  __bf16 bf16x8;
typedef __attribute__((ext_vector_type(16))) __bf16 bf16x16;

#define VOCAB 32000
#define D_IN  512
#define D_H   1024
#define BATCH 16
#define TSEQ  256
#define NBLK  64           // persistent recurrence blocks (one 16-col tile each)
#define SWX_STRIDE 516     // 512 + 4 : stride%64 == 4 -> conflict-free b64 reads
#define SWH_STRIDE 1028    // 1024 + 4

union BF16Pack { bf16x16 v; bf16x8 h[2]; };

// ---------------- utility kernels ----------------

__global__ void zero_f32(float* __restrict__ p, int n) {
    int i = blockIdx.x * blockDim.x + threadIdx.x;
    if (i < n) p[i] = 0.0f;
}

// fp32 -> bf16, 4 elements per thread
__global__ void cvt_bf16(const float* __restrict__ src, __bf16* __restrict__ dst, int n4) {
    int i = blockIdx.x * blockDim.x + threadIdx.x;
    if (i < n4) {
        f32x4 v = ((const f32x4*)src)[i];
        bf16x4 o;
        o[0] = (__bf16)v[0]; o[1] = (__bf16)v[1];
        o[2] = (__bf16)v[2]; o[3] = (__bf16)v[3];
        ((bf16x4*)dst)[i] = o;
    }
}

// ------------- persistent RNN recurrence (fp32 WMMA 16x16x4) -------------
// 64 single-wave blocks. Block b owns columns [16b, 16b+16) of h for ALL
// timesteps. Its Wx/Wh row-panels are staged in LDS ONCE (LDS survives the
// per-step acquire invalidate; weights never touch the global path again).
// Steps are separated by a device-scope arrive-counter barrier.
__global__ void __launch_bounds__(32)
rnn_persist(const int*   __restrict__ idx,
            const float* __restrict__ emb,
            const float* __restrict__ Wx,
            const float* __restrict__ Wh,
            const float* __restrict__ bias,
            float*       __restrict__ h0,
            float*       __restrict__ h1,
            __bf16*      __restrict__ h_all,
            unsigned int* __restrict__ counter)
{
    __shared__ float sWx[16 * SWX_STRIDE];   // 33,024 B
    __shared__ float sWh[16 * SWH_STRIDE];   // 65,792 B

    const int lane = threadIdx.x;       // 0..31
    const int lr   = lane & 15;
    const int half = lane >> 4;
    const int ko   = 2 * half;          // lane's K sub-offset for f32 WMMA operands
    const int j0   = blockIdx.x * 16;   // output column tile in H

    // ---- one-time weight staging: sWx[n][k] = Wx[j0+n][k], sWh[n][k] = Wh[j0+n][k]
    for (int i = lane; i < 16 * (D_IN / 4); i += 32) {       // 2048 f32x4
        const int row = i >> 7;                // 128 vec4 per row
        const int col = (i & 127) * 4;
        *(f32x4*)&sWx[row * SWX_STRIDE + col] =
            *(const f32x4*)(Wx + (long)(j0 + row) * D_IN + col);
    }
    for (int i = lane; i < 16 * (D_H / 4); i += 32) {        // 4096 f32x4
        const int row = i >> 8;                // 256 vec4 per row
        const int col = (i & 255) * 4;
        *(f32x4*)&sWh[row * SWH_STRIDE + col] =
            *(const f32x4*)(Wh + (long)(j0 + row) * D_H + col);
    }
    // single wave: LDS ops are in-order per wave, no barrier needed

    const float bv = bias[j0 + lr];
    const float* sbx = &sWx[lr * SWX_STRIDE + ko];   // B[k][n=lr] run for Wx
    const float* sbh = &sWh[lr * SWH_STRIDE + ko];   // B[k][n=lr] run for Wh

    for (int t = 0; t < TSEQ; ++t) {
        const float* hp = (t & 1) ? h1 : h0;
        float*       hn = (t & 1) ? h0 : h1;

        // A operand: row lr of x = emb[token]; pad token 0 hits the zero row.
        const int tokA = idx[lr * TSEQ + t];
        const float* arow = emb + (long)tokA * D_IN;

        f32x8 c = {0.f, 0.f, 0.f, 0.f, 0.f, 0.f, 0.f, 0.f};

        #pragma unroll 4
        for (int k = 0; k < D_IN; k += 4) {
            f32x2 a = *(const f32x2*)(arow + k + ko);
            f32x2 b = *(const f32x2*)(sbx + k);           // ds_load_b64, bank-clean
            c = __builtin_amdgcn_wmma_f32_16x16x4_f32(false, a, false, b,
                                                      (short)0, c, false, false);
        }

        const float* arow2 = hp + lr * D_H;
        #pragma unroll 4
        for (int k = 0; k < D_H; k += 4) {
            f32x2 a = *(const f32x2*)(arow2 + k + ko);
            f32x2 b = *(const f32x2*)(sbh + k);           // ds_load_b64, bank-clean
            c = __builtin_amdgcn_wmma_f32_16x16x4_f32(false, a, false, b,
                                                      (short)0, c, false, false);
        }

        // Epilogue: lane holds col n = lane%16, rows r + 8*(lane/16).
        #pragma unroll
        for (int r = 0; r < 8; ++r) {
            const int m   = r + 8 * half;          // batch row
            const int tok = idx[m * TSEQ + t];
            float hv = tanhf(c[r] + bv);
            if (tok == 0) hv = hp[m * D_H + j0 + lr];   // keep old h at pad
            hn[m * D_H + j0 + lr] = hv;
            h_all[((long)m * TSEQ + t) * D_H + j0 + lr] = (__bf16)hv;
        }

        // ---- grid-wide step barrier: release arrive, acquire spin ----
        if (lane == 0)
            __hip_atomic_fetch_add(counter, 1u, __ATOMIC_RELEASE,
                                   __HIP_MEMORY_SCOPE_AGENT);
        const unsigned int target = (unsigned int)NBLK * (unsigned int)(t + 1);
        while (__hip_atomic_load(counter, __ATOMIC_ACQUIRE,
                                 __HIP_MEMORY_SCOPE_AGENT) < target) {
            __builtin_amdgcn_s_sleep(1);
        }
    }
}

// ---------------- LM head GEMM (bf16 WMMA 16x16x32, fp32 accum) ----------------
// Block: 256 threads = 8 waves (2 M x 4 N), covering 64 rows x 256 cols.
// Wave: 32 rows x 64 cols = 8 accumulators; 12 b128 loads per 8 WMMAs.
__global__ void __launch_bounds__(256)
lm_head(const __bf16* __restrict__ A,    // [4096, 1024] row-major
        const __bf16* __restrict__ Bw,   // [32000, 1024] row-major (lm_W in bf16)
        float*        __restrict__ out)  // [4096, 32000]
{
    const int lane = threadIdx.x & 31;
    const int wave = threadIdx.x >> 5;   // 0..7
    const int wm   = wave >> 2;          // 0..1
    const int wn   = wave & 3;           // 0..3
    const int lr   = lane & 15;
    const int half = lane >> 4;

    const int m0 = blockIdx.y * 64 + wm * 32;    // wave owns rows [m0, m0+32)
    const int n0 = blockIdx.x * 256 + wn * 64;   // wave owns cols [n0, n0+64)

    const __bf16* ap[2];
    ap[0] = A + (long)(m0 + lr) * D_H;
    ap[1] = A + (long)(m0 + 16 + lr) * D_H;
    const __bf16* bp[4];
    #pragma unroll
    for (int j = 0; j < 4; ++j)
        bp[j] = Bw + (long)(n0 + 16 * j + lr) * D_H;

    f32x8 acc[2][4];
    #pragma unroll
    for (int i = 0; i < 2; ++i)
        #pragma unroll
        for (int j = 0; j < 4; ++j)
            acc[i][j] = (f32x8){0.f,0.f,0.f,0.f,0.f,0.f,0.f,0.f};

    #pragma unroll 2
    for (int k = 0; k < D_H; k += 32) {
        // A (16-bit, 16x32): lane holds K chunks [k+8h .. +7] and [k+16+8h .. +7]
        BF16Pack ua[2];
        #pragma unroll
        for (int i = 0; i < 2; ++i) {
            ua[i].h[0] = *(const bf16x8*)(ap[i] + k + 8 * half);
            ua[i].h[1] = *(const bf16x8*)(ap[i] + k + 16 + 8 * half);
        }
        // B (16-bit, 32x16): lane holds K chunks [k+8h .. +7] and [k+16+8h .. +7] of col lr
        BF16Pack ub[4];
        #pragma unroll
        for (int j = 0; j < 4; ++j) {
            ub[j].h[0] = *(const bf16x8*)(bp[j] + k + 8 * half);
            ub[j].h[1] = *(const bf16x8*)(bp[j] + k + 16 + 8 * half);
        }
        #pragma unroll
        for (int i = 0; i < 2; ++i)
            #pragma unroll
            for (int j = 0; j < 4; ++j)
                acc[i][j] = __builtin_amdgcn_wmma_f32_16x16x32_bf16(
                    false, ua[i].v, false, ub[j].v,
                    (short)0, acc[i][j], false, false);
    }

    #pragma unroll
    for (int i = 0; i < 2; ++i) {
        #pragma unroll
        for (int r = 0; r < 8; ++r) {
            const int m = m0 + 16 * i + r + 8 * half;
            float* orow = out + (long)m * VOCAB + n0 + lr;
            orow[ 0] = acc[i][0][r];
            orow[16] = acc[i][1][r];
            orow[32] = acc[i][2][r];
            orow[48] = acc[i][3][r];
        }
    }
}

// ---------------- launch ----------------

extern "C" void kernel_launch(void* const* d_in, const int* in_sizes, int n_in,
                              void* d_out, int out_size, void* d_ws, size_t ws_size,
                              hipStream_t stream)
{
    (void)in_sizes; (void)n_in; (void)out_size; (void)ws_size;

    const int*   idx  = (const int*)  d_in[0];   // [B, T]
    const float* emb  = (const float*)d_in[1];   // [V, D_IN]
    const float* Wx   = (const float*)d_in[2];   // [H, D_IN]
    const float* Wh   = (const float*)d_in[3];   // [H, H]
    const float* bias = (const float*)d_in[4];   // [H]
    const float* lmW  = (const float*)d_in[5];   // [V, H]
    float* out = (float*)d_out;                  // [B, T, V]

    char* ws = (char*)d_ws;
    const size_t HALL_BYTES = (size_t)BATCH * TSEQ * D_H * 2;   // 8 MB
    const size_t LMWB_BYTES = (size_t)VOCAB * D_H * 2;          // 64 MB
    __bf16* h_all = (__bf16*)ws;
    __bf16* lmWb  = (__bf16*)(ws + HALL_BYTES);
    float*  h0    = (float*)(ws + HALL_BYTES + LMWB_BYTES);
    float*  h1    = h0 + BATCH * D_H;
    unsigned int* counter = (unsigned int*)(h1 + BATCH * D_H);

    // 1) zero h ping-pong buffers + barrier counter (contiguous)
    {
        int n = 2 * BATCH * D_H + 1;
        zero_f32<<<dim3((n + 255) / 256), 256, 0, stream>>>(h0, n);
    }

    // 2) lm_W fp32 -> bf16 (64 MB; lives in L2 for the whole GEMM)
    {
        int n4 = VOCAB * D_H / 4;
        cvt_bf16<<<dim3((n4 + 255) / 256), 256, 0, stream>>>(lmW, lmWb, n4);
    }

    // 3) whole recurrence in ONE persistent kernel (counter barrier per step)
    rnn_persist<<<dim3(NBLK), 32, 0, stream>>>(idx, emb, Wx, Wh, bias,
                                               h0, h1, h_all, counter);

    // 4) LM head: [4096,1024] x [32000,1024]^T -> [4096,32000]
    lm_head<<<dim3(VOCAB / 256, (BATCH * TSEQ) / 64), 256, 0, stream>>>(h_all, lmWb, out);
}
